// Cross_AttentionHead_withMask_36429912605016
// MI455X (gfx1250) — matrix operand order repaired and checked by reading, not compiled, not run on hardware
//
#include <hip/hip_runtime.h>

#define BB   8
#define TQL  2048
#define TKL  4096
#define DIMG 512
#define DTXT 128
#define HS   64
#define QSCALE 0.04419417382415922f  /* 1/sqrt(512) */

typedef __attribute__((ext_vector_type(16))) _Float16 v16h;
typedef __attribute__((ext_vector_type(8)))  float    v8f;
typedef __attribute__((ext_vector_type(8)))  unsigned v8u;
typedef int v4i __attribute__((vector_size(16)));

union FragH { v16h h; v8u u; };

// ---- CDNA5 async global->LDS copy (ASYNCcnt-tracked), with sync fallback ----
#if __has_builtin(__builtin_amdgcn_global_load_async_to_lds_b128)
#define HAVE_ASYNC_LDS 1
#define ASYNC_COPY_B128(g, l)                                                  \
  __builtin_amdgcn_global_load_async_to_lds_b128(                              \
      (__attribute__((address_space(1))) v4i*)(g),                             \
      (__attribute__((address_space(3))) v4i*)(l), 0, 0)
#if __has_builtin(__builtin_amdgcn_s_wait_asynccnt)
#define ASYNC_WAIT(n) __builtin_amdgcn_s_wait_asynccnt(n)
#else
#define ASYNC_WAIT(n) asm volatile("s_wait_asynccnt %0" ::"i"(n))
#endif
#else
#define HAVE_ASYNC_LDS 0
#define ASYNC_COPY_B128(g, l)                                                  \
  do { *(uint4*)(l) = *(const uint4*)(g); } while (0)
#define ASYNC_WAIT(n) ((void)0)
#endif

__device__ __forceinline__ unsigned pack2h(float a, float b) {
  _Float16 ha = (_Float16)a, hb = (_Float16)b;
  unsigned short ua = __builtin_bit_cast(unsigned short, ha);
  unsigned short ub = __builtin_bit_cast(unsigned short, hb);
  return (unsigned)ua | ((unsigned)ub << 16);
}

// K index (even) of VGPR-pair p for this lane in a 16x32 f16 A-fragment
__device__ __forceinline__ int a_pair_kk(int lane, int p) {
  return ((lane >> 4) << 4) + (p << 1);
}

__device__ __forceinline__ v8f wmma_f16(const FragH& a, const FragH& b, v8f c) {
  return __builtin_amdgcn_wmma_f32_16x16x32_f16(false, a.h, false, b.h,
                                                (short)0, c, false, false);
}

// ---------------------------------------------------------------------------
// Repack W [HS, C] f32 -> [C/2, HS] u32 of packed (f16(kk), f16(kk+1)) pairs.
// ---------------------------------------------------------------------------
__global__ void pack_w_kernel(const float* __restrict__ W,
                              unsigned* __restrict__ Wp, int C) {
  int idx = blockIdx.x * blockDim.x + threadIdx.x;
  int total = (C >> 1) * HS;
  if (idx >= total) return;
  int pr = idx / HS;
  int h  = idx - pr * HS;
  int kk = pr << 1;
  Wp[idx] = pack2h(W[(size_t)h * C + kk], W[(size_t)h * C + kk + 1]);
}

// ---------------------------------------------------------------------------
// K/V projection + 2D RoPE on K. One wave per 16 rows of x_image.
//   Kp : [B*TK, 32] u32, row-major, head-dim pairs packed
//   Vtp: [B*64, TK/2] u32, V^T with key pairs packed
// ---------------------------------------------------------------------------
__global__ __launch_bounds__(128) void proj_kv_kernel(
    const float* __restrict__ xi, const unsigned* __restrict__ WkP,
    const unsigned* __restrict__ WvP, const float* __restrict__ fx,
    const float* __restrict__ fy, unsigned* __restrict__ Kp,
    unsigned* __restrict__ Vtp) {
  const int lane = threadIdx.x & 31;
  const int tile = blockIdx.x * (blockDim.x >> 5) + (threadIdx.x >> 5);
  const int b  = tile / (TKL / 16);
  const int t0 = (tile % (TKL / 16)) * 16;
  const int n  = lane & 15;
  const int m0 = (lane >> 4) << 3;

  v8f ck[4], cv[4];
  #pragma unroll
  for (int f = 0; f < 4; f++)
    #pragma unroll
    for (int p = 0; p < 8; p++) { ck[f][p] = 0.f; cv[f][p] = 0.f; }

  const float* arow = xi + ((size_t)(b * TKL + t0 + n)) * DIMG;
  for (int kk0 = 0; kk0 < DIMG; kk0 += 32) {
    // All independent loads first, then all WMMAs (lets the scheduler overlap)
    FragH af, bk[4], bv[4];
    #pragma unroll
    for (int p = 0; p < 8; p++) {
      float2 xv = *(const float2*)(arow + kk0 + a_pair_kk(lane, p));
      af.u[p] = pack2h(xv.x, xv.y);
    }
    const int pbase = (kk0 >> 1) + m0;
    #pragma unroll
    for (int f = 0; f < 4; f++)
      #pragma unroll
      for (int p = 0; p < 8; p++)
        bk[f].u[p] = WkP[(pbase + p) * HS + f * 16 + n];
    #pragma unroll
    for (int f = 0; f < 4; f++)
      #pragma unroll
      for (int p = 0; p < 8; p++)
        bv[f].u[p] = WvP[(pbase + p) * HS + f * 16 + n];
    #pragma unroll
    for (int f = 0; f < 4; f++) ck[f] = wmma_f16(af, bk[f], ck[f]);
    #pragma unroll
    for (int f = 0; f < 4; f++) cv[f] = wmma_f16(af, bv[f], cv[f]);
  }

  // 2D RoPE on K (dims 0..31 use fx, 32..63 use fy) + packed stores
  #pragma unroll
  for (int f = 0; f < 4; f++) {
    const int h = f * 16 + n;
    const float* ftab = (f < 2) ? fx : fy;
    const int i = (h & 31) >> 1;
    #pragma unroll
    for (int p = 0; p < 8; p++) {
      const int t = t0 + m0 + p;
      float2 cs = *(const float2*)(ftab + ((size_t)t * 16 + i) * 2);
      float a  = ck[f][p];
      float pr = __shfl_xor(a, 1, 32);
      float ov = (h & 1) ? (pr * cs.y + a * cs.x) : (a * cs.x - pr * cs.y);
      float po = __shfl_xor(ov, 1, 32);
      if ((h & 1) == 0)
        Kp[((size_t)(b * TKL + t)) * (HS / 2) + (h >> 1)] = pack2h(ov, po);
    }
    #pragma unroll
    for (int j = 0; j < 4; j++) {
      const int t = t0 + m0 + 2 * j;
      Vtp[((size_t)(b * HS + h)) * (TKL / 2) + (t >> 1)] =
          pack2h(cv[f][2 * j], cv[f][2 * j + 1]);
    }
  }
}

// ---------------------------------------------------------------------------
// Q projection + 1D RoPE + 1/sqrt(512) scale.  Qp: [B*TQ, 32] u32 pair-packed.
// ---------------------------------------------------------------------------
__global__ __launch_bounds__(128) void proj_q_kernel(
    const float* __restrict__ xt, const unsigned* __restrict__ WqP,
    const float* __restrict__ fl, unsigned* __restrict__ Qp) {
  const int lane = threadIdx.x & 31;
  const int tile = blockIdx.x * (blockDim.x >> 5) + (threadIdx.x >> 5);
  const int b  = tile / (TQL / 16);
  const int s0 = (tile % (TQL / 16)) * 16;
  const int n  = lane & 15;
  const int m0 = (lane >> 4) << 3;

  v8f cq[4];
  #pragma unroll
  for (int f = 0; f < 4; f++)
    #pragma unroll
    for (int p = 0; p < 8; p++) cq[f][p] = 0.f;

  const float* arow = xt + ((size_t)(b * TQL + s0 + n)) * DTXT;
  #pragma unroll
  for (int kk0 = 0; kk0 < DTXT; kk0 += 32) {
    FragH af, bq[4];
    #pragma unroll
    for (int p = 0; p < 8; p++) {
      float2 xv = *(const float2*)(arow + kk0 + a_pair_kk(lane, p));
      af.u[p] = pack2h(xv.x, xv.y);
    }
    const int pbase = (kk0 >> 1) + m0;
    #pragma unroll
    for (int f = 0; f < 4; f++)
      #pragma unroll
      for (int p = 0; p < 8; p++)
        bq[f].u[p] = WqP[(pbase + p) * HS + f * 16 + n];
    #pragma unroll
    for (int f = 0; f < 4; f++) cq[f] = wmma_f16(af, bq[f], cq[f]);
  }

  #pragma unroll
  for (int f = 0; f < 4; f++) {
    const int h = f * 16 + n;
    const int i = h >> 1;
    #pragma unroll
    for (int p = 0; p < 8; p++) {
      const int s = s0 + m0 + p;
      float2 cs = *(const float2*)(fl + ((size_t)s * 32 + i) * 2);
      float a  = cq[f][p];
      float pr = __shfl_xor(a, 1, 32);
      float ov = ((h & 1) ? (pr * cs.y + a * cs.x) : (a * cs.x - pr * cs.y)) * QSCALE;
      float po = __shfl_xor(ov, 1, 32);
      if ((h & 1) == 0)
        Qp[((size_t)(b * TQL + s)) * (HS / 2) + (h >> 1)] = pack2h(ov, po);
    }
  }
}

// ---------------------------------------------------------------------------
// Flash attention, 2 waves/block, one wave per 16 queries, 32 keys/step.
// K and V tiles are double-buffered in wave-private LDS and filled with
// GLOBAL_LOAD_ASYNC_TO_LDS_B128 (ASYNCcnt), overlapping the previous step's
// WMMAs.  S^T = K x Q^T, online softmax, P via LDS, out^T += V^T x P.
// ---------------------------------------------------------------------------
#define KTW  (32 * 33)   /* padded K tile, u32 */
#define VTW  (64 * 17)   /* padded V tile, u32 */
#define WAVE_LDS (2 * KTW + 2 * VTW + 256)

__device__ __forceinline__ void stage_k_tile(const unsigned* gbase,
                                             unsigned* lds, int lane) {
  // 32 rows x 32 u32, contiguous in global; LDS row stride 33
  #pragma unroll
  for (int i = 0; i < 8; i++) {
    int tr = i * 4 + (lane >> 3);
    int cb = (lane & 7) * 4;
    ASYNC_COPY_B128(gbase + tr * 32 + cb, lds + tr * 33 + cb);
  }
}

__device__ __forceinline__ void stage_v_tile(const unsigned* gbase,
                                             unsigned* lds, int lane) {
  // 64 rows (head dims) x 16 u32 (key pairs); LDS row stride 17
  #pragma unroll
  for (int i = 0; i < 8; i++) {
    int h  = i * 8 + (lane >> 2);
    int cb = (lane & 3) * 4;
    ASYNC_COPY_B128(gbase + (size_t)h * (TKL / 2) + cb, lds + h * 17 + cb);
  }
}

__global__ __launch_bounds__(64) void attn_kernel(
    const unsigned* __restrict__ Qp, const unsigned* __restrict__ Kp,
    const unsigned* __restrict__ Vtp, float* __restrict__ out) {
  __shared__ unsigned smem[2 * WAVE_LDS];
  unsigned* wbase = smem + (threadIdx.x >> 5) * WAVE_LDS;
  unsigned* klds[2] = {wbase, wbase + KTW};
  unsigned* vlds[2] = {wbase + 2 * KTW, wbase + 2 * KTW + VTW};
  unsigned* pl = wbase + 2 * KTW + 2 * VTW;

  const int lane = threadIdx.x & 31;
  const int w = blockIdx.x * (blockDim.x >> 5) + (threadIdx.x >> 5);
  const int b  = w / (TQL / 16);
  const int s0 = (w % (TQL / 16)) * 16;
  const int q  = lane & 15;
  const int m0 = (lane >> 4) << 3;

  const unsigned* kg = Kp + (size_t)b * TKL * (HS / 2);
  const unsigned* vg = Vtp + (size_t)b * HS * (TKL / 2);

  // Q^T B-fragments, loaded once
  FragH qb[2];
  {
    const unsigned* qrow = Qp + ((size_t)(b * TQL + s0 + q)) * (HS / 2);
    #pragma unroll
    for (int half = 0; half < 2; half++)
      #pragma unroll
      for (int p = 0; p < 8; p++)
        qb[half].u[p] = qrow[half * 16 + m0 + p];
  }

  v8f acc[4];
  #pragma unroll
  for (int f = 0; f < 4; f++)
    #pragma unroll
    for (int p = 0; p < 8; p++) acc[f][p] = 0.f;
  float mrun = -3.0e38f, lrun = 0.f;

  // Prologue: stage chunk 0 (16 async transfers)
  stage_k_tile(kg, klds[0], lane);
  stage_v_tile(vg, vlds[0], lane);

  for (int t0 = 0; t0 < TKL; t0 += 32) {
    const int cur = (t0 >> 5) & 1;
    if (t0 + 32 < TKL) {
      stage_k_tile(kg + (size_t)(t0 + 32) * (HS / 2), klds[cur ^ 1], lane);
      stage_v_tile(vg + ((t0 + 32) >> 1), vlds[cur ^ 1], lane);
      ASYNC_WAIT(16);  // current chunk's 16 transfers complete (in-order)
    } else {
      ASYNC_WAIT(0);
    }
    const unsigned* kc = klds[cur];
    const unsigned* vc = vlds[cur];

    // K A-fragments from LDS: 2 key tiles x 2 head-dim halves
    FragH ka[2][2];
    #pragma unroll
    for (int kt = 0; kt < 2; kt++) {
      const unsigned* kl = kc + (kt * 16 + q) * 33;
      #pragma unroll
      for (int half = 0; half < 2; half++)
        #pragma unroll
        for (int p = 0; p < 8; p++)
          ka[kt][half].u[p] = kl[(half * 32 + a_pair_kk(lane, p)) >> 1];
    }
    v8f c0, c1;
    #pragma unroll
    for (int p = 0; p < 8; p++) { c0[p] = 0.f; c1[p] = 0.f; }
    c0 = wmma_f16(ka[0][0], qb[0], c0);
    c0 = wmma_f16(ka[0][1], qb[1], c0);
    c1 = wmma_f16(ka[1][0], qb[0], c1);
    c1 = wmma_f16(ka[1][1], qb[1], c1);

    // Online softmax over key columns (lane pair q, q+16 covers all 32 keys)
    float mloc = -3.0e38f;
    #pragma unroll
    for (int p = 0; p < 8; p++) mloc = fmaxf(mloc, fmaxf(c0[p], c1[p]));
    float mchunk = fmaxf(mloc, __shfl_xor(mloc, 16, 32));
    float mnew = fmaxf(mrun, mchunk);
    float alpha = __expf(mrun - mnew);
    float e0[8], e1[8], lloc = 0.f;
    #pragma unroll
    for (int p = 0; p < 8; p++) {
      e0[p] = __expf(c0[p] - mnew);
      e1[p] = __expf(c1[p] - mnew);
      lloc += e0[p] + e1[p];
    }
    lrun = lrun * alpha + lloc + __shfl_xor(lloc, 16, 32);
    mrun = mnew;
    #pragma unroll
    for (int f = 0; f < 4; f++)
      #pragma unroll
      for (int p = 0; p < 8; p++) acc[f][p] *= alpha;

    // Stage P (32 keys x 16 queries, key pairs packed) in wave-private LDS
    #pragma unroll
    for (int j = 0; j < 4; j++) {
      pl[((m0 >> 1) + j) * 16 + q]     = pack2h(e0[2 * j], e0[2 * j + 1]);
      pl[(8 + (m0 >> 1) + j) * 16 + q] = pack2h(e1[2 * j], e1[2 * j + 1]);
    }
    FragH pb;
    #pragma unroll
    for (int p = 0; p < 8; p++) pb.u[p] = pl[(m0 + p) * 16 + q];

    // out^T += V^T x P over the 4 head-dim tiles (V from LDS)
    FragH va[4];
    #pragma unroll
    for (int f = 0; f < 4; f++) {
      const unsigned* vl = vc + (f * 16 + q) * 17;
      #pragma unroll
      for (int p = 0; p < 8; p++)
        va[f].u[p] = vl[a_pair_kk(lane, p) >> 1];
    }
    #pragma unroll
    for (int f = 0; f < 4; f++) acc[f] = wmma_f16(va[f], pb, acc[f]);
  }

  float inv = 1.f / lrun;
  #pragma unroll
  for (int f = 0; f < 4; f++)
    #pragma unroll
    for (int p = 0; p < 8; p++)
      out[((size_t)(b * TQL + s0 + q)) * HS + f * 16 + m0 + p] = acc[f][p] * inv;
}

// ---------------------------------------------------------------------------
extern "C" void kernel_launch(void* const* d_in, const int* in_sizes, int n_in,
                              void* d_out, int out_size, void* d_ws, size_t ws_size,
                              hipStream_t stream) {
  (void)in_sizes; (void)n_in; (void)out_size; (void)ws_size;
  const float* x_image = (const float*)d_in[0];
  const float* x_text  = (const float*)d_in[1];
  /* d_in[2] (x_latex_mask) is unused by the reference */
  const float* f_lat   = (const float*)d_in[3];
  const float* f_imx   = (const float*)d_in[4];
  const float* f_imy   = (const float*)d_in[5];
  const float* Wk      = (const float*)d_in[6];
  const float* Wq      = (const float*)d_in[7];
  const float* Wv      = (const float*)d_in[8];
  float* out = (float*)d_out;

  unsigned* ws  = (unsigned*)d_ws;
  unsigned* WkP = ws;                                // 16384 u32
  unsigned* WqP = WkP + (DIMG / 2) * HS;             //  4096 u32
  unsigned* WvP = WqP + (DTXT / 2) * HS;             // 16384 u32
  unsigned* Kp  = WvP + (DIMG / 2) * HS;             // B*TK*32   u32 (4 MB)
  unsigned* Vtp = Kp + (size_t)BB * TKL * (HS / 2);  // B*64*TK/2 u32 (4 MB)
  unsigned* Qp  = Vtp + (size_t)BB * HS * (TKL / 2); // B*TQ*32   u32 (2 MB)

  pack_w_kernel<<<((DIMG / 2) * HS + 255) / 256, 256, 0, stream>>>(Wk, WkP, DIMG);
  pack_w_kernel<<<((DTXT / 2) * HS + 255) / 256, 256, 0, stream>>>(Wq, WqP, DTXT);
  pack_w_kernel<<<((DIMG / 2) * HS + 255) / 256, 256, 0, stream>>>(Wv, WvP, DIMG);

  proj_kv_kernel<<<(BB * TKL / 16) / 4, 128, 0, stream>>>(
      x_image, WkP, WvP, f_imx, f_imy, Kp, Vtp);
  proj_q_kernel<<<(BB * TQL / 16) / 4, 128, 0, stream>>>(x_text, WqP, f_lat, Qp);
  attn_kernel<<<(BB * (TQL / 16)) / 2, 64, 0, stream>>>(Qp, Kp, Vtp, out);
}
